// AdditiveAttention_44616120271031
// MI455X (gfx1250) — matrix operand order repaired and checked by reading, not compiled, not run on hardware
//
#include <hip/hip_runtime.h>
#include <hip/hip_bf16.h>
#include <math.h>

// Problem constants (match reference)
#define B_   8
#define LQ_  128
#define LK_  1024
#define D_   512      // DQ == DK == DV
#define H_   256
#define NEG_ (-1000000.0f)
#define KC   32       // keys per flash chunk (== WMMA K for bf16)

typedef __bf16 bf16_t;
typedef __attribute__((ext_vector_type(16))) __bf16 v16bf;
typedef __attribute__((ext_vector_type(8)))  float  v8f;

// ---- fast tanh: prefer builtin (v_tanh_f32), else inline asm, never OCML ----
__device__ __forceinline__ float tanh_fast(float x) {
#if __has_builtin(__builtin_amdgcn_tanhf)
    return __builtin_amdgcn_tanhf(x);
#elif __has_builtin(__builtin_amdgcn_tanh_f32)
    return __builtin_amdgcn_tanh_f32(x);
#else
    float r;
    asm("v_tanh_f32 %0, %1" : "=v"(r) : "v"(x));
    return r;
#endif
}

// ---- async global->LDS staging (gfx1250 ASYNCcnt path), guarded fallback ----
#if __has_builtin(__builtin_amdgcn_global_load_async_to_lds_b128) && \
    __has_builtin(__builtin_amdgcn_s_wait_asynccnt)
#define HAVE_ASYNC_LDS 1
#endif

#ifdef HAVE_ASYNC_LDS
typedef int v4i __attribute__((vector_size(16)));
typedef __attribute__((address_space(1))) v4i gv4i;   // global (__device__) AS
typedef __attribute__((address_space(3))) v4i lv4i;   // LDS (__shared__) AS
#endif

__device__ __forceinline__ void copy16_g2lds(float* lds_dst, const float* gsrc) {
#ifdef HAVE_ASYNC_LDS
    __builtin_amdgcn_global_load_async_to_lds_b128(
        (gv4i*)gsrc, (lv4i*)lds_dst, 0, 0);
#else
    *(float4*)lds_dst = *(const float4*)gsrc;
#endif
}
__device__ __forceinline__ void async_copy_fence() {
#ifdef HAVE_ASYNC_LDS
    __builtin_amdgcn_s_wait_asynccnt(0);
#endif
}

__device__ __forceinline__ bf16_t cvt_bf16(float x) { return (bf16_t)x; }

// -------------------------------------------------------------------------
// P[M,H] = X[M,D] @ W[D,H] via v_wmma_f32_16x16x32_bf16.
// One wave per 16x16 output tile; 8 waves (256 threads) per block.
// Fragment k order: contiguous, k = 16*(lane/16) + s, same for A and B (wave32).
// -------------------------------------------------------------------------
__global__ __launch_bounds__(256) void proj_wmma_kernel(
    const float* __restrict__ X, const float* __restrict__ W,
    float* __restrict__ P, int M, int D, int Hh)
{
    const int lane = threadIdx.x & 31;
    const int wave = threadIdx.x >> 5;
    const int tile = blockIdx.x * 8 + wave;
    const int ntn  = Hh >> 4;
    const int tm   = tile / ntn;
    const int tn   = tile % ntn;
    if (tm * 16 >= M) return;                 // wave-uniform

    const int hi   = lane >> 4;
    const int l15  = lane & 15;
    const int mrow = tm * 16 + l15;
    const int ncol = tn * 16 + l15;
    const float* __restrict__ xrow = X + (size_t)mrow * D;

    v8f c = {};
    for (int k0 = 0; k0 < D; k0 += 32) {
        v16bf a, b;
#pragma unroll
        for (int j = 0; j < 4; ++j) {
            const int ka  = k0 + 16 * hi + 2 * j;
            const int ka2 = k0 + 16 * hi + 8 + 2 * j;
            a[2*j]     = cvt_bf16(xrow[ka]);
            a[2*j + 1] = cvt_bf16(xrow[ka + 1]);
            a[8 + 2*j] = cvt_bf16(xrow[ka2]);
            a[9 + 2*j] = cvt_bf16(xrow[ka2 + 1]);
        }
#pragma unroll
        for (int j = 0; j < 8; ++j) {
            const int kb = k0 + 16 * hi + 2 * j;
            b[2*j]     = cvt_bf16(W[(size_t)kb * Hh + ncol]);
            b[2*j + 1] = cvt_bf16(W[(size_t)(kb + 1) * Hh + ncol]);
        }
        c = __builtin_amdgcn_wmma_f32_16x16x32_bf16(
                false, a, false, b, (short)0, c, false, false);
    }
#pragma unroll
    for (int i = 0; i < 8; ++i)
        P[(size_t)(tm * 16 + i + 8 * hi) * Hh + ncol] = c[i];
}

// -------------------------------------------------------------------------
// Fused additive-attention: scores(tanh) + masked online softmax + P@V.
// One 256-thread block per (batch b, 16-query tile qt). Flash-style single
// pass over LK in KC=32 key chunks. Wave w accumulates DV tiles
// {w, w+8, w+16, w+24} in fp32 WMMA accumulators.
// -------------------------------------------------------------------------
__global__ __launch_bounds__(256) void addattn_kernel(
    const float* __restrict__ qp,   // [B, LQ, H]
    const float* __restrict__ kp,   // [B, LK, H]
    const float* __restrict__ V,    // [B, LK, D]
    const float* __restrict__ wv,   // [H]
    const int*   __restrict__ valid_lens,  // [B]
    float* __restrict__ out)        // [B, LQ, D]
{
    __shared__ alignas(16) float qp_s[16][H_ + 4];  // stride 260: 16B-aligned rows,
    __shared__ alignas(16) float kp_s[KC][H_ + 4];  // conflict-free (260 % 64 == 4)
    __shared__ alignas(16) float wv_s[H_];
    __shared__ alignas(16) float sc_s[16][KC + 1];
    __shared__ float scale_s[16];

    const int tid  = threadIdx.x;
    const int lane = tid & 31;
    const int wave = tid >> 5;
    const int hi   = lane >> 4;
    const int l15  = lane & 15;
    const int b    = blockIdx.x >> 3;    // LQ/16 == 8 query tiles
    const int qt   = blockIdx.x & 7;

    const int vlen = valid_lens[b];

    // Stage query projections for this tile (contiguous 16*H floats, async).
    const float* __restrict__ qsrc = qp + (size_t)(b * LQ_ + qt * 16) * H_;
    for (int i = tid * 4; i < 16 * H_; i += 256 * 4)
        copy16_g2lds(&qp_s[i >> 8][i & (H_ - 1)], qsrc + i);
    wv_s[tid] = wv[tid];                 // H_ == 256 == blockDim

    // Flash softmax state; rows owned by threads 0..15.
    float m_run = -3.0e38f, l_run = 0.0f;
    v8f acc[4] = {};

    for (int kc = 0; kc < LK_; kc += KC) {
        __syncthreads();                 // kp_s no longer being read
        const float* __restrict__ ksrc = kp + (size_t)(b * LK_ + kc) * H_;
        for (int i = tid * 4; i < KC * H_; i += 256 * 4)
            copy16_g2lds(&kp_s[i >> 8][i & (H_ - 1)], ksrc + i);
        async_copy_fence();              // this wave's async LDS writes done
        __syncthreads();                 // all waves' staging visible

        // scores: s[q][k] = sum_h wv[h]*tanh(qp[q][h] + kp[k][h])  (masked)
        // Each thread: one key, two query rows -> kv/wv reads amortized.
        {
            const int k  = tid >> 3;          // 0..31
            const int q0 = (tid & 7) * 2;     // 0,2,...,14
            const int q1 = q0 + 1;
            float s0 = 0.0f, s1 = 0.0f;
            for (int h = 0; h < H_; h += 4) {
                const float4 kv = *(const float4*)&kp_s[k][h];
                const float4 qa = *(const float4*)&qp_s[q0][h];
                const float4 qb = *(const float4*)&qp_s[q1][h];
                const float4 wr = *(const float4*)&wv_s[h];
                s0 += wr.x * tanh_fast(qa.x + kv.x);
                s0 += wr.y * tanh_fast(qa.y + kv.y);
                s0 += wr.z * tanh_fast(qa.z + kv.z);
                s0 += wr.w * tanh_fast(qa.w + kv.w);
                s1 += wr.x * tanh_fast(qb.x + kv.x);
                s1 += wr.y * tanh_fast(qb.y + kv.y);
                s1 += wr.z * tanh_fast(qb.z + kv.z);
                s1 += wr.w * tanh_fast(qb.w + kv.w);
            }
            const bool valid = (kc + k) < vlen;
            sc_s[q0][k] = valid ? s0 : NEG_;
            sc_s[q1][k] = valid ? s1 : NEG_;
        }
        __syncthreads();

        // Online softmax update, one thread per query row.
        if (tid < 16) {
            float mc = -3.0e38f;
#pragma unroll
            for (int k = 0; k < KC; ++k) mc = fmaxf(mc, sc_s[tid][k]);
            const float mnew  = fmaxf(m_run, mc);
            const float scale = __expf(m_run - mnew);
            float psum = 0.0f;
#pragma unroll
            for (int k = 0; k < KC; ++k) {
                const float pv = __expf(sc_s[tid][k] - mnew);
                sc_s[tid][k] = pv;       // overwrite with exp weights
                psum += pv;
            }
            l_run = l_run * scale + psum;
            m_run = mnew;
            scale_s[tid] = scale;
        }
        __syncthreads();

        // Rescale accumulators by per-row scale (row = i + 8*hi in C layout).
        float rs[8];
#pragma unroll
        for (int i = 0; i < 8; ++i) rs[i] = scale_s[8 * hi + i];
#pragma unroll
        for (int t = 0; t < 4; ++t)
#pragma unroll
            for (int i = 0; i < 8; ++i) acc[t][i] *= rs[i];

        // A fragment: exp weights P (16 x KC) from LDS, bf16.
        v16bf a;
#pragma unroll
        for (int j = 0; j < 4; ++j) {
            a[2*j]     = cvt_bf16(sc_s[l15][16 * hi + 2 * j]);
            a[2*j + 1] = cvt_bf16(sc_s[l15][16 * hi + 2 * j + 1]);
            a[8 + 2*j] = cvt_bf16(sc_s[l15][16 * hi + 8 + 2 * j]);
            a[9 + 2*j] = cvt_bf16(sc_s[l15][16 * hi + 8 + 2 * j + 1]);
        }
        // B fragment per DV tile: V chunk (KC x 16), coalesced along l15.
        const float* __restrict__ vbase =
            V + (size_t)(b * LK_ + kc + 16 * hi) * D_ + l15;
#pragma unroll
        for (int t = 0; t < 4; ++t) {
            const int dvb = (wave + 8 * t) * 16;
            v16bf bb;
#pragma unroll
            for (int j = 0; j < 8; ++j) {
                bb[2*j]     = cvt_bf16(vbase[(size_t)(2 * j) * D_ + dvb]);
                bb[2*j + 1] = cvt_bf16(vbase[(size_t)(2 * j + 1) * D_ + dvb]);
            }
            acc[t] = __builtin_amdgcn_wmma_f32_16x16x32_bf16(
                         false, a, false, bb, (short)0, acc[t], false, false);
        }
    }

    // Finalize: divide by softmax denominator, store.
    __syncthreads();
    if (tid < 16) scale_s[tid] = 1.0f / l_run;
    __syncthreads();
    float inv[8];
#pragma unroll
    for (int i = 0; i < 8; ++i) inv[i] = scale_s[8 * hi + i];
#pragma unroll
    for (int t = 0; t < 4; ++t) {
        const int dvb = (wave + 8 * t) * 16;
#pragma unroll
        for (int i = 0; i < 8; ++i) {
            out[(size_t)(b * LQ_ + qt * 16 + i + 8 * hi) * D_ + dvb + l15] =
                acc[t][i] * inv[i];
        }
    }
}

// -------------------------------------------------------------------------
extern "C" void kernel_launch(void* const* d_in, const int* in_sizes, int n_in,
                              void* d_out, int out_size, void* d_ws, size_t ws_size,
                              hipStream_t stream) {
    (void)in_sizes; (void)n_in; (void)out_size; (void)ws_size;
    const float* queries = (const float*)d_in[0];  // [B,LQ,D]
    const float* keys    = (const float*)d_in[1];  // [B,LK,D]
    const float* values  = (const float*)d_in[2];  // [B,LK,D]
    const float* Wq      = (const float*)d_in[3];  // [D,H]
    const float* Wk      = (const float*)d_in[4];  // [D,H]
    const float* wv      = (const float*)d_in[5];  // [H]
    const int*   vlens   = (const int*)d_in[6];    // [B]
    float* out = (float*)d_out;

    float* qp = (float*)d_ws;                      // [B*LQ, H]  = 1 MB
    float* kp = qp + (size_t)B_ * LQ_ * H_;        // [B*LK, H]  = 8 MB

    // qp = queries @ Wq : 1024 tiles -> 128 blocks of 8 waves
    proj_wmma_kernel<<<(B_ * LQ_ / 16) * (H_ / 16) / 8, 256, 0, stream>>>(
        queries, Wq, qp, B_ * LQ_, D_, H_);
    // kp = keys @ Wk : 8192 tiles -> 1024 blocks
    proj_wmma_kernel<<<(B_ * LK_ / 16) * (H_ / 16) / 8, 256, 0, stream>>>(
        keys, Wk, kp, B_ * LK_, D_, H_);
    // Fused scores + masked softmax + P@V : one block per (b, q-tile)
    addattn_kernel<<<B_ * (LQ_ / 16), 256, 0, stream>>>(
        qp, kp, values, wv, vlens, out);
}
